// GNNLayer_60619168416174
// MI455X (gfx1250) — matrix operand order repaired and checked by reading, not compiled, not run on hardware
//
#include <hip/hip_runtime.h>

typedef _Float16 half_t;
typedef __attribute__((ext_vector_type(16))) _Float16 v16h;
typedef __attribute__((ext_vector_type(8)))  float    v8f;

#define NN 50000
#define NE 800000
#define DD 128

// ---------------------------------------------------------------- helpers

__device__ __forceinline__ v8f wmma_f16(v16h a, v16h b, v8f c) {
    // D = A(16x32 f16) * B(32x16 f16) + C(16x16 f32)
    return __builtin_amdgcn_wmma_f32_16x16x32_f16(
        /*neg_a=*/false, a, /*neg_b=*/false, b,
        /*c_mod=*/(short)0, c, /*reuse_a=*/false, /*reuse_b=*/false);
}

__device__ __forceinline__ v16h load_frag2(const half_t* p0, const half_t* p1) {
    v16h r;
    uint4* rp = reinterpret_cast<uint4*>(&r);
    rp[0] = *reinterpret_cast<const uint4*>(p0);   // halves 0..7
    rp[1] = *reinterpret_cast<const uint4*>(p1);   // halves 8..15
    return r;
}

__device__ __forceinline__ v16h load_bfrag(const half_t* __restrict__ base, int idx) {
    const uint4* p = reinterpret_cast<const uint4*>(base) + (size_t)idx * 2;
    v16h r;
    uint4* rp = reinterpret_cast<uint4*>(&r);
    rp[0] = p[0];
    rp[1] = p[1];
    return r;
}

__device__ __forceinline__ void atomic_add_f32(float* addr, float val) {
    asm volatile("global_atomic_add_f32 %0, %1, off"
                 : : "v"(addr), "v"(val) : "memory");
}

// ---------------------------------------------------------------- prep kernels

// Pack weight W [K=KB*32, 128] (row-major, fp32) into per-lane B fragments (f16).
// Fragment index = (n_tile * KB + kb) * 32 + lane; 16 halves per lane:
//   lanes 0-15 : K = kb*32 + 0..7, 16..23,  col = n_tile*16 + lane
//   lanes 16-31: K = kb*32 + 8..15, 24..31, col = n_tile*16 + (lane-16)
__global__ void pack_w_kernel(const float* __restrict__ W, half_t* __restrict__ outw, int KB) {
    int tid = blockIdx.x * blockDim.x + threadIdx.x;
    int total = 8 * KB * 32;
    if (tid >= total) return;
    int lane = tid & 31;
    int kb   = (tid >> 5) % KB;
    int t    = (tid >> 5) / KB;
    int k0   = kb * 32 + ((lane >= 16) ? 8 : 0);
    int j    = t * 16 + (lane & 15);
    half_t* o = outw + (size_t)tid * 16;
#pragma unroll
    for (int i = 0; i < 16; ++i)
        o[i] = (half_t)W[(size_t)(k0 + (i < 8 ? i : i + 8)) * DD + j];
}

__global__ void cvt_x_kernel(const float* __restrict__ x, half_t* __restrict__ xh) {
    size_t i = (size_t)(blockIdx.x * blockDim.x + threadIdx.x) * 8;
    __align__(16) half_t tmp[8];
#pragma unroll
    for (int j = 0; j < 8; ++j) tmp[j] = (half_t)x[i + j];
    *reinterpret_cast<uint4*>(xh + i) = *reinterpret_cast<const uint4*>(tmp);
}

__global__ void zero_kernel(float4* __restrict__ p) {
    size_t i = (size_t)(blockIdx.x * blockDim.x + threadIdx.x);
    p[i] = make_float4(0.f, 0.f, 0.f, 0.f);
}

// ---------------------------------------------------------------- edge kernel
// One block = 16 edges; 8 waves each own a 16-column slice of the 128 outputs.

__global__ __launch_bounds__(256) void edge_msg_kernel(
    const half_t* __restrict__ xh, const int* __restrict__ ei,
    const half_t* __restrict__ wB1, const float* __restrict__ b1,
    const half_t* __restrict__ wB2, const float* __restrict__ b2,
    float* __restrict__ agg)
{
    __shared__ __align__(16) half_t tileA[16][264];  // 16 edges x 256 feats (padded)
    __shared__ __align__(16) half_t tileH[16][136];  // 16 edges x 128 hidden (padded)
    __shared__ int s_row[16];
    __shared__ int s_col[16];

    const int tid  = threadIdx.x;
    const int lane = tid & 31;
    const int wave = tid >> 5;
    const int e0   = blockIdx.x * 16;

    if (tid < 16) {
        s_row[tid] = ei[e0 + tid];
        s_col[tid] = ei[NE + e0 + tid];
    }
    __syncthreads();

    {   // stage concat(x[row], x[col]) -> tileA, coalesced b128 loads
        const int m = tid >> 4;
        const int c = tid & 15;
        const uint4* sr = reinterpret_cast<const uint4*>(xh + (size_t)s_row[m] * DD);
        const uint4* sc = reinterpret_cast<const uint4*>(xh + (size_t)s_col[m] * DD);
        *reinterpret_cast<uint4*>(&tileA[m][c * 8])       = sr[c];
        *reinterpret_cast<uint4*>(&tileA[m][DD + c * 8])  = sc[c];
    }
    __syncthreads();

    const int mrow = lane & 15;
    const int off  = (lane >= 16) ? 8 : 0;   // A-fragment K sub-offset / C-row offset
    const int jl   = lane & 15;              // local output column

    // ---- layer 1: h = relu(A[16x256] * mW1 + b1), this wave's 16 columns
    v8f acc = {0.f, 0.f, 0.f, 0.f, 0.f, 0.f, 0.f, 0.f};
#pragma unroll
    for (int kb = 0; kb < 8; ++kb) {
        const int k0 = kb * 32;
        v16h a = load_frag2(&tileA[mrow][k0 + off], &tileA[mrow][k0 + 16 + off]);
        v16h b = load_bfrag(wB1, (wave * 8 + kb) * 32 + lane);
        acc = wmma_f16(a, b, acc);
    }
    {
        const float bb = b1[wave * 16 + jl];
#pragma unroll
        for (int r = 0; r < 8; ++r) {
            float v = acc[r] + bb;
            v = v > 0.f ? v : 0.f;
            tileH[r + off][wave * 16 + jl] = (half_t)v;
        }
    }
    __syncthreads();

    // ---- layer 2: msg = h[16x128] * mW2 + b2
    v8f acc2 = {0.f, 0.f, 0.f, 0.f, 0.f, 0.f, 0.f, 0.f};
#pragma unroll
    for (int kb = 0; kb < 4; ++kb) {
        const int k0 = kb * 32;
        v16h a = load_frag2(&tileH[mrow][k0 + off], &tileH[mrow][k0 + 16 + off]);
        v16h b = load_bfrag(wB2, (wave * 4 + kb) * 32 + lane);
        acc2 = wmma_f16(a, b, acc2);
    }
    {
        const float bb = b2[wave * 16 + jl];
        const int j = wave * 16 + jl;
#pragma unroll
        for (int r = 0; r < 8; ++r) {
            float v = acc2[r] + bb;
            atomic_add_f32(agg + (size_t)s_col[r + off] * DD + j, v);
        }
    }
}

// ---------------------------------------------------------------- node kernel
// One block = 16 nodes; upd = relu(relu([x|agg]*uW1+b1)*uW2+b2)

__global__ __launch_bounds__(256) void node_upd_kernel(
    const half_t* __restrict__ xh, const float* __restrict__ agg,
    const half_t* __restrict__ wB1, const float* __restrict__ b1,
    const half_t* __restrict__ wB2, const float* __restrict__ b2,
    float* __restrict__ out)
{
    __shared__ __align__(16) half_t tileA[16][264];
    __shared__ __align__(16) half_t tileG[16][136];

    const int tid  = threadIdx.x;
    const int lane = tid & 31;
    const int wave = tid >> 5;
    const int v0   = blockIdx.x * 16;

    {   // stage concat(x[v], agg[v]) -> tileA
        const int m = tid >> 4;
        const int c = tid & 15;
        const int v = v0 + m;
        const uint4* sx = reinterpret_cast<const uint4*>(xh + (size_t)v * DD);
        *reinterpret_cast<uint4*>(&tileA[m][c * 8]) = sx[c];
        const float* ap = agg + (size_t)v * DD + c * 8;
        __align__(16) half_t tmp[8];
#pragma unroll
        for (int i = 0; i < 8; ++i) tmp[i] = (half_t)ap[i];
        *reinterpret_cast<uint4*>(&tileA[m][DD + c * 8]) =
            *reinterpret_cast<const uint4*>(tmp);
    }
    __syncthreads();

    const int mrow = lane & 15;
    const int off  = (lane >= 16) ? 8 : 0;
    const int jl   = lane & 15;

    v8f acc = {0.f, 0.f, 0.f, 0.f, 0.f, 0.f, 0.f, 0.f};
#pragma unroll
    for (int kb = 0; kb < 8; ++kb) {
        const int k0 = kb * 32;
        v16h a = load_frag2(&tileA[mrow][k0 + off], &tileA[mrow][k0 + 16 + off]);
        v16h b = load_bfrag(wB1, (wave * 8 + kb) * 32 + lane);
        acc = wmma_f16(a, b, acc);
    }
    {
        const float bb = b1[wave * 16 + jl];
#pragma unroll
        for (int r = 0; r < 8; ++r) {
            float v = acc[r] + bb;
            v = v > 0.f ? v : 0.f;
            tileG[r + off][wave * 16 + jl] = (half_t)v;
        }
    }
    __syncthreads();

    v8f acc2 = {0.f, 0.f, 0.f, 0.f, 0.f, 0.f, 0.f, 0.f};
#pragma unroll
    for (int kb = 0; kb < 4; ++kb) {
        const int k0 = kb * 32;
        v16h a = load_frag2(&tileG[mrow][k0 + off], &tileG[mrow][k0 + 16 + off]);
        v16h b = load_bfrag(wB2, (wave * 4 + kb) * 32 + lane);
        acc2 = wmma_f16(a, b, acc2);
    }
    {
        const float bb = b2[wave * 16 + jl];
        const int j = wave * 16 + jl;
#pragma unroll
        for (int r = 0; r < 8; ++r) {
            float v = acc2[r] + bb;
            v = v > 0.f ? v : 0.f;
            out[(size_t)(v0 + r + off) * DD + j] = v;
        }
    }
}

// ---------------------------------------------------------------- launch

extern "C" void kernel_launch(void* const* d_in, const int* in_sizes, int n_in,
                              void* d_out, int out_size, void* d_ws, size_t ws_size,
                              hipStream_t stream) {
    const float* x   = (const float*)d_in[0];
    const int*   ei  = (const int*)d_in[1];
    const float* mW1 = (const float*)d_in[2];
    const float* mb1 = (const float*)d_in[3];
    const float* mW2 = (const float*)d_in[4];
    const float* mb2 = (const float*)d_in[5];
    const float* uW1 = (const float*)d_in[6];
    const float* ub1 = (const float*)d_in[7];
    const float* uW2 = (const float*)d_in[8];
    const float* ub2 = (const float*)d_in[9];
    float* out = (float*)d_out;
    char*  ws  = (char*)d_ws;

    // workspace layout (bytes)
    half_t* wB1m = (half_t*)(ws + 0);         // 8*8*32*16 halves  = 65536 B
    half_t* wB2m = (half_t*)(ws + 65536);     // 8*4*32*16 halves  = 32768 B
    half_t* wB1u = (half_t*)(ws + 98304);     // 65536 B
    half_t* wB2u = (half_t*)(ws + 163840);    // 32768 B
    half_t* xh   = (half_t*)(ws + 196608);    // N*128*2 = 12800000 B
    float*  agg  = (float*)(ws + 12996608);   // N*128*4 = 25600000 B

    // 1) pack weights into B-fragment order (f16)
    pack_w_kernel<<<8, 256, 0, stream>>>(mW1, wB1m, 8);
    pack_w_kernel<<<4, 256, 0, stream>>>(mW2, wB2m, 4);
    pack_w_kernel<<<8, 256, 0, stream>>>(uW1, wB1u, 8);
    pack_w_kernel<<<4, 256, 0, stream>>>(uW2, wB2u, 4);

    // 2) x -> f16 (N*D/8 threads), 3) zero aggregation buffer (N*D/4 float4s)
    cvt_x_kernel<<<(NN * DD / 8) / 256, 256, 0, stream>>>(x, xh);
    zero_kernel<<<(NN * DD / 4) / 256, 256, 0, stream>>>((float4*)agg);

    // 4) edge MLP + scatter-add (E/16 blocks, exact)
    edge_msg_kernel<<<NE / 16, 256, 0, stream>>>(xh, ei, wB1m, mb1, wB2m, mb2, agg);

    // 5) node MLP (N/16 blocks, exact)
    node_upd_kernel<<<NN / 16, 256, 0, stream>>>(xh, agg, wB1u, ub1, wB2u, ub2, out);
}
